// SelfAttention_42898133353131
// MI455X (gfx1250) — compile-verified, matrix operand order checked by reading, not run on hardware
//
#include <hip/hip_runtime.h>

// ---------------- problem constants ----------------
constexpr int BATCH   = 2;
constexpr int TLEN    = 1024;
constexpr int EMB     = 1024;
constexpr int NH      = 16;
constexpr int DH      = 64;
constexpr int MEMLEN  = 1024;
constexpr int LMEMLEN = 128;
constexpr int KVLEN   = TLEN + MEMLEN + LMEMLEN;   // 2176
constexpr int TOTMEM  = MEMLEN + LMEMLEN;          // 1152
constexpr float ATT_SCALE = 0.125f;                // 64^-0.5

// ---------------- WMMA types/helpers ----------------
typedef __attribute__((ext_vector_type(16))) __bf16 v16bf;
typedef __attribute__((ext_vector_type(8)))  float  v8f;

struct __align__(16) U128 { unsigned int x, y, z, w; };
union FragB16 { v16bf v; U128 q[2]; unsigned int u[8]; };

__device__ __forceinline__ v8f wmma_bf16(v16bf a, v16bf b, v8f c) {
    return __builtin_amdgcn_wmma_f32_16x16x32_bf16(false, a, false, b,
                                                   (short)0, c, false, false);
}

// Load a 16x32 bf16 fragment (A or B^T layout) from a row-major [16][>=32] tile.
// Per CDNA5 ISA 16-bit A layout: lane covers matrix row (lane&15);
// lanes 0-15 hold K = {0..7,16..23}, lanes 16-31 hold K = {8..15,24..31}.
__device__ __forceinline__ v16bf load_frag(const __bf16* tile, int rowStride, int lane) {
    const __bf16* r = tile + (size_t)(lane & 15) * rowStride;
    int g = (lane >> 4) & 1;
    FragB16 f;
    f.q[0] = *(const U128*)(r + g * 8);
    f.q[1] = *(const U128*)(r + 16 + g * 8);
    return f.v;
}

// Same, but row index = jrow + (lane&15) into pos-emb table, zero-filled when OOB.
__device__ __forceinline__ v16bf load_frag_pe(const __bf16* pe_h, int jrow, int kbase, int lane) {
    int j = jrow + (lane & 15);
    FragB16 f;
    if (j >= 0 && j < KVLEN) {
        const __bf16* r = pe_h + (size_t)j * DH + kbase;
        int g = (lane >> 4) & 1;
        f.q[0] = *(const U128*)(r + g * 8);
        f.q[1] = *(const U128*)(r + 16 + g * 8);
    } else {
        for (int i = 0; i < 8; i++) f.u[i] = 0u;
    }
    return f.v;
}

// 32x32 C macro-tile per wave: 2x2 WMMA tiles, each A/B fragment reused twice.
// A: row-major [32][lda] starting at Arow; Bt: row-major [32][ldb] (transposed weights).
__device__ __forceinline__ void gemm32x32(const __bf16* __restrict__ A,
                                          const __bf16* __restrict__ Bt,
                                          int lda, int ldb, int K, int lane,
                                          v8f& c00, v8f& c01, v8f& c10, v8f& c11) {
    const __bf16* A1 = A + (size_t)16 * lda;
    const __bf16* B1 = Bt + (size_t)16 * ldb;
    for (int k = 0; k < K; k += 32) {
        v16bf a0 = load_frag(A + k, lda, lane);
        v16bf a1 = load_frag(A1 + k, lda, lane);
        v16bf b0 = load_frag(Bt + k, ldb, lane);
        v16bf b1 = load_frag(B1 + k, ldb, lane);
        c00 = wmma_bf16(a0, b0, c00);
        c01 = wmma_bf16(a0, b1, c01);
        c10 = wmma_bf16(a1, b0, c10);
        c11 = wmma_bf16(a1, b1, c11);
    }
}

// ---------------- pack / convert kernels ----------------
__global__ void cvt_kernel(const float* __restrict__ s, __bf16* __restrict__ d, long long n) {
    long long stride = (long long)gridDim.x * blockDim.x;
    for (long long i = (long long)blockIdx.x * blockDim.x + threadIdx.x; i < n; i += stride)
        d[i] = (__bf16)s[i];
}

// dst[c*R + r] = src[r*C + c]  (transpose + convert; dst is [C][R])
__global__ void cvt_t_kernel(const float* __restrict__ s, __bf16* __restrict__ d, int R, int C) {
    long long n = (long long)R * C;
    long long stride = (long long)gridDim.x * blockDim.x;
    for (long long i = (long long)blockIdx.x * blockDim.x + threadIdx.x; i < n; i += stride) {
        long long c = i / R, r = i - c * R;
        d[i] = (__bf16)s[r * (long long)C + c];
    }
}

// kv_input = concat([lmem, mem, x], axis=1) -> bf16 [B][KVLEN][EMB]
__global__ void concat_kv_kernel(const float* __restrict__ x, const float* __restrict__ mem,
                                 const float* __restrict__ lmem, __bf16* __restrict__ kvc) {
    long long n = (long long)BATCH * KVLEN * EMB;
    long long stride = (long long)gridDim.x * blockDim.x;
    for (long long i = (long long)blockIdx.x * blockDim.x + threadIdx.x; i < n; i += stride) {
        int e = (int)(i % EMB);
        long long rem = i / EMB;
        int j = (int)(rem % KVLEN);
        int b = (int)(rem / KVLEN);
        float v;
        if (j < LMEMLEN)
            v = lmem[((size_t)b * LMEMLEN + j) * EMB + e];
        else if (j < LMEMLEN + MEMLEN)
            v = mem[((size_t)b * MEMLEN + (j - LMEMLEN)) * EMB + e];
        else
            v = x[((size_t)b * TLEN + (j - TOTMEM)) * EMB + e];
        kvc[i] = (__bf16)v;
    }
}

// ---------------- projection GEMMs (WMMA, 32x32 macro-tile per wave) ----------------
__global__ __launch_bounds__(128)
void gemm_q_kernel(const __bf16* __restrict__ Xb, const __bf16* __restrict__ WqT,
                   __bf16* __restrict__ Qb) {
    const int lane = threadIdx.x & 31, wave = threadIdx.x >> 5;
    const int b = blockIdx.z;
    const int n0 = blockIdx.x * 32;
    const int m0 = (blockIdx.y * 4 + wave) * 32;
    v8f c00 = {}, c01 = {}, c10 = {}, c11 = {};
    gemm32x32(Xb + ((size_t)b * TLEN + m0) * EMB, WqT + (size_t)n0 * EMB,
              EMB, EMB, EMB, lane, c00, c01, c10, c11);
    const int half = lane >> 4, col = lane & 15;
    v8f* accs[4] = { &c00, &c01, &c10, &c11 };
    for (int t = 0; t < 4; t++) {
        int mi = (t >> 1) * 16, nj = (t & 1) * 16;
        int c = n0 + nj + col;
        int h = c >> 6, d = c & 63;
        for (int r = 0; r < 8; r++) {
            int m = m0 + mi + r + 8 * half;
            Qb[(((size_t)b * NH + h) * TLEN + m) * DH + d] = (__bf16)((*accs[t])[r]);
        }
    }
}

__global__ __launch_bounds__(128)
void gemm_kv_kernel(const __bf16* __restrict__ KVc, const __bf16* __restrict__ WkvT,
                    __bf16* __restrict__ Kb, __bf16* __restrict__ Vt) {
    const int lane = threadIdx.x & 31, wave = threadIdx.x >> 5;
    const int b = blockIdx.z;
    const int n0 = blockIdx.x * 32;
    const int m0 = (blockIdx.y * 4 + wave) * 32;   // kv position tile
    v8f c00 = {}, c01 = {}, c10 = {}, c11 = {};
    gemm32x32(KVc + ((size_t)b * KVLEN + m0) * EMB, WkvT + (size_t)n0 * EMB,
              EMB, EMB, EMB, lane, c00, c01, c10, c11);
    const int half = lane >> 4, col = lane & 15;
    v8f* accs[4] = { &c00, &c01, &c10, &c11 };
    for (int t = 0; t < 4; t++) {
        int mi = (t >> 1) * 16, nj = (t & 1) * 16;
        int c = n0 + nj + col;
        for (int r = 0; r < 8; r++) {
            int n = m0 + mi + r + 8 * half;
            float v = (*accs[t])[r];
            if (c < EMB) {  // K half: [b,h,kv,dh]
                int h = c >> 6, d = c & 63;
                Kb[(((size_t)b * NH + h) * KVLEN + n) * DH + d] = (__bf16)v;
            } else {        // V half: stored transposed [b,h,dh,kv]
                int c2 = c - EMB;
                int h = c2 >> 6, d = c2 & 63;
                Vt[(((size_t)b * NH + h) * DH + d) * KVLEN + n] = (__bf16)v;
            }
        }
    }
}

__global__ __launch_bounds__(128)
void gemm_out_kernel(const __bf16* __restrict__ Ob, const __bf16* __restrict__ WoutT,
                     const float* __restrict__ bout, float* __restrict__ out) {
    const int lane = threadIdx.x & 31, wave = threadIdx.x >> 5;
    const int b = blockIdx.z;
    const int n0 = blockIdx.x * 32;
    const int m0 = (blockIdx.y * 4 + wave) * 32;
    v8f c00 = {}, c01 = {}, c10 = {}, c11 = {};
    gemm32x32(Ob + ((size_t)b * TLEN + m0) * EMB, WoutT + (size_t)n0 * EMB,
              EMB, EMB, EMB, lane, c00, c01, c10, c11);
    const int half = lane >> 4, col = lane & 15;
    v8f* accs[4] = { &c00, &c01, &c10, &c11 };
    for (int t = 0; t < 4; t++) {
        int mi = (t >> 1) * 16, nj = (t & 1) * 16;
        int c = n0 + nj + col;
        float bias = bout[c];
        for (int r = 0; r < 8; r++) {
            int m = m0 + mi + r + 8 * half;
            out[((size_t)b * TLEN + m) * EMB + c] = (*accs[t])[r] + bias;
        }
    }
}

// ---------------- fused rel-pos flash attention ----------------
// One wave handles 16 query rows (m0..m0+15) of one (b,h); iterates kv in 32-key steps.
// Scores: S = Q K^T (WMMA) + diagonal-gathered pos window P = Q PE^T (WMMA via LDS).
__global__ __launch_bounds__(256)
void attn_kernel(const __bf16* __restrict__ Qb, const __bf16* __restrict__ Kb,
                 const __bf16* __restrict__ Vt, const __bf16* __restrict__ PEb,
                 __bf16* __restrict__ Ob) {
    __shared__ __align__(16) float  posw[8][16 * 48];   // 24 KB
    __shared__ __align__(16) __bf16 pstage[8][16 * 32]; // 8 KB

    const int lane = threadIdx.x & 31, wave = threadIdx.x >> 5;
    const int bh = blockIdx.x;
    const int bb = bh >> 4, hh = bh & 15;
    const int m0 = (blockIdx.y * 8 + wave) * 16;
    const int half = lane >> 4;
    const int col  = lane & 15;

    const __bf16* qbase  = Qb + (((size_t)bb * NH + hh) * TLEN + m0) * DH;
    const __bf16* kbase  = Kb + (((size_t)bb * NH + hh) * KVLEN) * DH;
    const __bf16* vbase  = Vt + (((size_t)bb * NH + hh) * DH) * KVLEN;
    const __bf16* pebase = PEb + (size_t)hh * KVLEN * DH;

    v16bf qa0 = load_frag(qbase, DH, lane);        // d = 0..31
    v16bf qa1 = load_frag(qbase + 32, DH, lane);   // d = 32..63

    v8f o0 = {}, o1 = {}, o2 = {}, o3 = {};
    float rmax[8], rsum[8];
    for (int r = 0; r < 8; r++) { rmax[r] = -3.0e38f; rsum[r] = 0.0f; }

    int nlimit = m0 + 15 + TOTMEM + 1;           // last valid key + 1
    if (nlimit > KVLEN) nlimit = KVLEN;
    const int nend = (nlimit + 31) & ~31;

    float*  pw = &posw[wave][0];
    __bf16* ps = &pstage[wave][0];

    for (int n0 = 0; n0 < nend; n0 += 32) {
        __builtin_prefetch(kbase + (size_t)(n0 + 32) * DH, 0, 1);

        // ---- content scores S = Q K^T ----
        v8f s0 = {}, s1 = {};
        {
            const __bf16* k0 = kbase + (size_t)n0 * DH;
            s0 = wmma_bf16(qa0, load_frag(k0, DH, lane), s0);
            s0 = wmma_bf16(qa1, load_frag(k0 + 32, DH, lane), s0);
            const __bf16* k1 = kbase + (size_t)(n0 + 16) * DH;
            s1 = wmma_bf16(qa0, load_frag(k1, DH, lane), s1);
            s1 = wmma_bf16(qa1, load_frag(k1 + 32, DH, lane), s1);
        }

        // ---- positional window: P[mi,c] = q[mi]·pe[jb+c], c in [0,48) ----
        const int jb = n0 + (TLEN - 1) - m0 - 15;
        for (int w = 0; w < 3; w++) {
            v8f p = {};
            p = wmma_bf16(qa0, load_frag_pe(pebase, jb + w * 16, 0, lane), p);
            p = wmma_bf16(qa1, load_frag_pe(pebase, jb + w * 16, 32, lane), p);
            for (int r = 0; r < 8; r++)
                pw[(r + 8 * half) * 48 + w * 16 + col] = p[r];
        }

        // ---- combine + causal mask + online softmax ----
        float p0[8], p1[8], curmax[8];
        for (int r = 0; r < 8; r++) {
            int row = r + 8 * half;
            int m = m0 + row;
            int nA = n0 + col;
            int nB = n0 + 16 + col;
            // pos term index: j = n + t-1 - m  ->  window col = (n - n0) - row + 15
            float sa = (nA <= m + TOTMEM)
                ? ATT_SCALE * (s0[r] + pw[row * 48 + (col - row + 15)]) : -3.0e38f;
            float sb = (nB <= m + TOTMEM)
                ? ATT_SCALE * (s1[r] + pw[row * 48 + (col - row + 31)]) : -3.0e38f;
            p0[r] = sa; p1[r] = sb;
            curmax[r] = fmaxf(sa, sb);
        }
        for (int off = 1; off < 16; off <<= 1)
            for (int r = 0; r < 8; r++)
                curmax[r] = fmaxf(curmax[r], __shfl_xor(curmax[r], off, 32));

        float alpha[8];
        for (int r = 0; r < 8; r++) {
            float nm = fmaxf(rmax[r], curmax[r]);
            alpha[r] = __expf(rmax[r] - nm);
            rmax[r] = nm;
            p0[r] = (p0[r] > -1.0e38f) ? __expf(p0[r] - nm) : 0.0f;
            p1[r] = (p1[r] > -1.0e38f) ? __expf(p1[r] - nm) : 0.0f;
        }
        float psum[8];
        for (int r = 0; r < 8; r++) psum[r] = p0[r] + p1[r];
        for (int off = 1; off < 16; off <<= 1)
            for (int r = 0; r < 8; r++)
                psum[r] += __shfl_xor(psum[r], off, 32);

        for (int r = 0; r < 8; r++) {
            rsum[r] = rsum[r] * alpha[r] + psum[r];
            o0[r] *= alpha[r]; o1[r] *= alpha[r];
            o2[r] *= alpha[r]; o3[r] *= alpha[r];
            int row = r + 8 * half;
            ps[row * 32 + col]      = (__bf16)p0[r];   // C-layout -> LDS
            ps[row * 32 + 16 + col] = (__bf16)p1[r];
        }

        // ---- O += P @ V  (re-fragment P from LDS as A-matrix) ----
        v16bf pa = load_frag(ps, 32, lane);
        o0 = wmma_bf16(pa, load_frag(vbase + (size_t) 0 * KVLEN + n0, KVLEN, lane), o0);
        o1 = wmma_bf16(pa, load_frag(vbase + (size_t)16 * KVLEN + n0, KVLEN, lane), o1);
        o2 = wmma_bf16(pa, load_frag(vbase + (size_t)32 * KVLEN + n0, KVLEN, lane), o2);
        o3 = wmma_bf16(pa, load_frag(vbase + (size_t)48 * KVLEN + n0, KVLEN, lane), o3);
    }

    // ---- normalize + write merged-head output [b][t][e] ----
    __bf16* obase = Ob + ((size_t)bb * TLEN + m0) * EMB + hh * 64;
    for (int r = 0; r < 8; r++) {
        int row = r + 8 * half;
        float inv = 1.0f / rsum[r];
        obase[(size_t)row * EMB +  0 + col] = (__bf16)(o0[r] * inv);
        obase[(size_t)row * EMB + 16 + col] = (__bf16)(o1[r] * inv);
        obase[(size_t)row * EMB + 32 + col] = (__bf16)(o2[r] * inv);
        obase[(size_t)row * EMB + 48 + col] = (__bf16)(o3[r] * inv);
    }
}

// ---------------- host launcher ----------------
extern "C" void kernel_launch(void* const* d_in, const int* in_sizes, int n_in,
                              void* d_out, int out_size, void* d_ws, size_t ws_size,
                              hipStream_t stream) {
    const float* x        = (const float*)d_in[0];
    const float* mem      = (const float*)d_in[1];
    const float* lmem     = (const float*)d_in[2];
    const float* pos_emb  = (const float*)d_in[3];
    // d_in[4] = input_mask (all true in this problem; pad mask is a no-op)
    const float* Wq       = (const float*)d_in[5];
    const float* Wkv      = (const float*)d_in[6];
    const float* Wout     = (const float*)d_in[7];
    const float* bout     = (const float*)d_in[8];
    float* out = (float*)d_out;

    char* w = (char*)d_ws;
    size_t off = 0;
    auto alloc = [&](size_t elems) -> __bf16* {
        __bf16* p = (__bf16*)(w + off);
        off = (off + elems * 2 + 255) & ~(size_t)255;
        return p;
    };
    __bf16* WqT   = alloc((size_t)EMB * EMB);
    __bf16* WkvT  = alloc((size_t)2 * EMB * EMB);
    __bf16* WoutT = alloc((size_t)EMB * EMB);
    __bf16* PEb   = alloc((size_t)NH * KVLEN * DH);
    __bf16* Xb    = alloc((size_t)BATCH * TLEN * EMB);
    __bf16* KVc   = alloc((size_t)BATCH * KVLEN * EMB);
    __bf16* Qb    = alloc((size_t)BATCH * NH * TLEN * DH);
    __bf16* Kb    = alloc((size_t)BATCH * NH * KVLEN * DH);
    __bf16* Vt    = alloc((size_t)BATCH * NH * DH * KVLEN);
    __bf16* Ob    = alloc((size_t)BATCH * TLEN * EMB);

    // pack / convert
    cvt_t_kernel<<<1024, 256, 0, stream>>>(Wq,   WqT,   EMB, EMB);
    cvt_t_kernel<<<2048, 256, 0, stream>>>(Wkv,  WkvT,  EMB, 2 * EMB);
    cvt_t_kernel<<<1024, 256, 0, stream>>>(Wout, WoutT, EMB, EMB);
    cvt_kernel  <<<2048, 256, 0, stream>>>(pos_emb, PEb, (long long)NH * KVLEN * DH);
    cvt_kernel  <<<2048, 256, 0, stream>>>(x, Xb, (long long)BATCH * TLEN * EMB);
    concat_kv_kernel<<<4096, 256, 0, stream>>>(x, mem, lmem, KVc);

    // projections (32x32 macro-tile per wave, 4 waves per block)
    gemm_q_kernel <<<dim3(EMB / 32,     TLEN / 128,  BATCH), 128, 0, stream>>>(Xb, WqT, Qb);
    gemm_kv_kernel<<<dim3(2 * EMB / 32, KVLEN / 128, BATCH), 128, 0, stream>>>(KVc, WkvT, Kb, Vt);

    // fused rel-pos attention
    attn_kernel<<<dim3(BATCH * NH, TLEN / 128), 256, 0, stream>>>(Qb, Kb, Vt, PEb, Ob);

    // output projection + bias
    gemm_out_kernel<<<dim3(EMB / 32, TLEN / 128, BATCH), 128, 0, stream>>>(Ob, WoutT, bout, out);
    (void)in_sizes; (void)n_in; (void)out_size; (void)ws_size;
}
